// MultiHeadAttention_24386824307211
// MI455X (gfx1250) — matrix operand order repaired and checked by reading, not compiled, not run on hardware
//
#include <hip/hip_runtime.h>

#define CB 4
#define CS 1024
#define CD 1024
#define CH 16
#define CDK 64

typedef __bf16 bf16;
typedef __attribute__((ext_vector_type(16))) __bf16 v16bf;
typedef __attribute__((ext_vector_type(8)))  __bf16 v8bf;
typedef __attribute__((ext_vector_type(4)))  __bf16 v4bf;
typedef __attribute__((ext_vector_type(8)))  float  v8f;

static __device__ __forceinline__ v8f wmma_bf16(v16bf a, v16bf b, v8f c) {
  // D = A(16x32 bf16) * B(32x16 bf16) + C(16x16 f32)
  return __builtin_amdgcn_wmma_f32_16x16x32_bf16(false, a, false, b, (short)0, c,
                                                 false, false);
}

// A-fragment (16x32 bf16) from a row-major row pointer.
// Lane holds row m = lane&15, h = lane>>4.
// VGPR i<4 packs k = {2i+8h, 2i+1+8h}; VGPR 4+j packs k = {16+2j+8h, 17+2j+8h}.
// => two contiguous 16B loads at k-offsets kbase+8h and kbase+16+8h.
static __device__ __forceinline__ v16bf load_afrag(const bf16* __restrict__ row,
                                                   int kbase, int h) {
  v8bf lo = *(const v8bf*)(row + kbase + 8 * h);
  v8bf hi = *(const v8bf*)(row + kbase + 16 + 8 * h);
  return __builtin_shufflevector(lo, hi, 0, 1, 2, 3, 4, 5, 6, 7,
                                 8, 9, 10, 11, 12, 13, 14, 15);
}

// B-fragment (32x16 bf16): lane holds col n = lane&15; k = 8h + (0..7), 16 + 8h + (0..7).
// Source row must have consecutive k contiguous (i.e. a row of B^T).
static __device__ __forceinline__ v16bf load_bfrag(const bf16* __restrict__ row,
                                                   int kbase, int h) {
  v8bf lo = *(const v8bf*)(row + kbase + 8 * h);
  v8bf hi = *(const v8bf*)(row + kbase + 16 + 8 * h);
  return __builtin_shufflevector(lo, hi, 0, 1, 2, 3, 4, 5, 6, 7,
                                 8, 9, 10, 11, 12, 13, 14, 15);
}

// ---------------------------------------------------------------- cvt kernel
__global__ void __launch_bounds__(256)
mha_cvt_kernel(const float* __restrict__ in, bf16* __restrict__ out, int n4) {
  int i = blockIdx.x * blockDim.x + threadIdx.x;
  if (i < n4) {
    float4 v = ((const float4*)in)[i];
    v4bf o = { (bf16)v.x, (bf16)v.y, (bf16)v.z, (bf16)v.w };
    ((v4bf*)out)[i] = o;
  }
}

// --------------------------------------------------------- QKV projection
// y = x @ W^T + b. Each wave computes a 32x64 output tile (2 M-subtiles x
// 4 N-subtiles): 8 WMMAs per K-step on 12 b128 loads (fragment reuse).
// sel 0->Q, 1->K (layout [B,H,S,DK]), 2->V (transposed [B,H,DK,S])
__global__ void __launch_bounds__(128)
mha_qkv_proj_kernel(const bf16* __restrict__ xq, const bf16* __restrict__ xk,
                    const bf16* __restrict__ xv,
                    const bf16* __restrict__ wq, const bf16* __restrict__ wk,
                    const bf16* __restrict__ wv,
                    const float* __restrict__ bq, const float* __restrict__ bk,
                    const float* __restrict__ bv,
                    bf16* __restrict__ Qp, bf16* __restrict__ Kp,
                    bf16* __restrict__ Vt) {
  const int lane = threadIdx.x & 31;
  const int wave = threadIdx.x >> 5;
  const int h4   = lane >> 4;
  const int ln   = lane & 15;

  const int mt  = blockIdx.x;               // (B*S)/32 = 128 tiles of 32 rows
  const int nq  = blockIdx.y * 4 + wave;    // D/64 = 16 quads of 16 cols
  const int sel = blockIdx.z;

  const bf16*  x    = sel == 0 ? xq : sel == 1 ? xk : xv;
  const bf16*  wmat = sel == 0 ? wq : sel == 1 ? wk : wv;
  const float* bias = sel == 0 ? bq : sel == 1 ? bk : bv;

  const bf16* arow0 = x + (size_t)(mt * 32 + ln) * CD;
  const bf16* arow1 = x + (size_t)(mt * 32 + 16 + ln) * CD;
  const bf16* brow0 = wmat + (size_t)(nq * 64 + 0  + ln) * CD;
  const bf16* brow1 = wmat + (size_t)(nq * 64 + 16 + ln) * CD;
  const bf16* brow2 = wmat + (size_t)(nq * 64 + 32 + ln) * CD;
  const bf16* brow3 = wmat + (size_t)(nq * 64 + 48 + ln) * CD;

  v8f acc[2][4] = {};
#pragma unroll 2
  for (int k = 0; k < CD; k += 32) {
    v16bf a0 = load_afrag(arow0, k, h4);
    v16bf a1 = load_afrag(arow1, k, h4);
    v16bf b0 = load_bfrag(brow0, k, h4);
    v16bf b1 = load_bfrag(brow1, k, h4);
    v16bf b2 = load_bfrag(brow2, k, h4);
    v16bf b3 = load_bfrag(brow3, k, h4);
    acc[0][0] = wmma_bf16(a0, b0, acc[0][0]);
    acc[0][1] = wmma_bf16(a0, b1, acc[0][1]);
    acc[0][2] = wmma_bf16(a0, b2, acc[0][2]);
    acc[0][3] = wmma_bf16(a0, b3, acc[0][3]);
    acc[1][0] = wmma_bf16(a1, b0, acc[1][0]);
    acc[1][1] = wmma_bf16(a1, b1, acc[1][1]);
    acc[1][2] = wmma_bf16(a1, b2, acc[1][2]);
    acc[1][3] = wmma_bf16(a1, b3, acc[1][3]);
  }

#pragma unroll
  for (int j = 0; j < 4; j++) {
    const int ng = nq * 64 + j * 16 + ln;
    const float bb = bias[ng];
    const int hh = ng >> 6, dk = ng & 63;
#pragma unroll
    for (int i = 0; i < 2; i++) {
#pragma unroll
      for (int r = 0; r < 8; r++) {
        int mg = mt * 32 + i * 16 + r + 8 * h4;
        int bidx = mg >> 10, s = mg & (CS - 1);
        bf16 vb = (bf16)(acc[i][j][r] + bb);
        if (sel == 2) {
          Vt[(((size_t)bidx * CH + hh) * CDK + dk) * CS + s] = vb;
        } else {
          bf16* dst = sel == 0 ? Qp : Kp;
          dst[(((size_t)bidx * CH + hh) * CS + s) * CDK + dk] = vb;
        }
      }
    }
  }
}

// ------------------------------------------------------------- attention
// One wave = 16 query rows; online-softmax over key chunks of 32.
// Scores computed transposed (S^T = K * Q^T) so the P fragment for P*V is a
// pure per-lane f32->bf16 repack of the score D-fragments (no LDS transpose).
__global__ void __launch_bounds__(128)
mha_attn_kernel(const bf16* __restrict__ Qp, const bf16* __restrict__ Kp,
                const bf16* __restrict__ Vt, const int* __restrict__ mask,
                bf16* __restrict__ xo) {
  const int lane = threadIdx.x & 31;
  const int wave = threadIdx.x >> 5;
  const int h4   = lane >> 4;
  const int ln   = lane & 15;

  const int hh = blockIdx.y, bb = blockIdx.z;
  const int q0 = blockIdx.x * 64 + wave * 16;

  const bf16* Qh  = Qp + ((size_t)bb * CH + hh) * CS * CDK;
  const bf16* Kh  = Kp + ((size_t)bb * CH + hh) * CS * CDK;
  const bf16* Vth = Vt + ((size_t)bb * CH + hh) * CDK * CS;
  const int* mrow = mask + ((size_t)bb * CS + (q0 + ln)) * CS;  // this lane's query row

  // Q as B-fragment: B[d][q] = Q[q0+q, d]; two K-steps (d 0..31, 32..63)
  const bf16* qrow = Qh + (size_t)(q0 + ln) * CDK;
  const v16bf qb0 = load_bfrag(qrow, 0, h4);
  const v16bf qb1 = load_bfrag(qrow, 32, h4);

  v8f o0 = {}, o1 = {}, o2 = {}, o3 = {};
  float row_max = -3.0e38f, row_sum = 0.0f;

  for (int jj = 0; jj < CS; jj += 32) {
    const bf16* krow0 = Kh + (size_t)(jj + ln) * CDK;       // A rows: keys jj+0..15
    const bf16* krow1 = Kh + (size_t)(jj + 16 + ln) * CDK;  // keys jj+16..31

    if (jj + 32 < CS) {  // prefetch next key chunk (128B rows -> 1 line each)
      __builtin_prefetch(Kh + (size_t)(jj + 32 + ln) * CDK, 0, 1);
      __builtin_prefetch(Kh + (size_t)(jj + 48 + ln) * CDK, 0, 1);
    }

    v8f s0 = {}, s1 = {};
    s0 = wmma_bf16(load_afrag(krow0, 0, h4),  qb0, s0);
    s0 = wmma_bf16(load_afrag(krow0, 32, h4), qb1, s0);
    s1 = wmma_bf16(load_afrag(krow1, 0, h4),  qb0, s1);
    s1 = wmma_bf16(load_afrag(krow1, 32, h4), qb1, s1);

    // mask values for this lane: keys jj+8h+0..7 and jj+16+8h+0..7 are
    // contiguous ints -> four b128 loads instead of 16 scalar loads.
    int m0[8], m1[8];
    *(int4*)&m0[0] = *(const int4*)(mrow + jj + 8 * h4);
    *(int4*)&m0[4] = *(const int4*)(mrow + jj + 8 * h4 + 4);
    *(int4*)&m1[0] = *(const int4*)(mrow + jj + 16 + 8 * h4);
    *(int4*)&m1[4] = *(const int4*)(mrow + jj + 16 + 8 * h4 + 4);

    // scale 1/sqrt(64), mask (-1e9 where mask==0), chunk max
    float cmax = -3.0e38f;
#pragma unroll
    for (int r = 0; r < 8; r++) {
      float v0 = (m0[r] == 0) ? -1.0e9f : s0[r] * 0.125f;
      float v1 = (m1[r] == 0) ? -1.0e9f : s1[r] * 0.125f;
      s0[r] = v0; s1[r] = v1;
      cmax = fmaxf(cmax, fmaxf(v0, v1));
    }
    cmax = fmaxf(cmax, __shfl_xor(cmax, 16, 32));  // join key halves

    float nmax  = fmaxf(row_max, cmax);
    float alpha = __expf(row_max - nmax);
    float psum  = 0.0f;

    // P = exp(s - nmax), repacked into A-fragment layout (per-lane only):
    // pa[0..7] <- tile0 regs, pa[8..15] <- tile1 regs.
    v16bf pa;
#pragma unroll
    for (int r = 0; r < 8; r++) {
      float p0 = __expf(s0[r] - nmax);
      float p1 = __expf(s1[r] - nmax);
      psum += p0 + p1;
      pa[r]     = (bf16)p0;
      pa[r + 8] = (bf16)p1;
    }
    psum += __shfl_xor(psum, 16, 32);
    row_sum = row_sum * alpha + psum;
    row_max = nmax;

    // rescale O accumulators: row m = r + 8*h4 needs alpha of that query,
    // which lives in lane (r + 8*h4) (query == lane&15 there).
#pragma unroll
    for (int r = 0; r < 8; r++) {
      float ar = __shfl(alpha, r + 8 * h4, 32);
      o0[r] *= ar; o1[r] *= ar; o2[r] *= ar; o3[r] *= ar;
    }

    // O += P(16x32) @ V(32x64); B-fragments from V^T rows (contiguous keys)
    o0 = wmma_bf16(pa, load_bfrag(Vth + (size_t)(0  + ln) * CS, jj, h4), o0);
    o1 = wmma_bf16(pa, load_bfrag(Vth + (size_t)(16 + ln) * CS, jj, h4), o1);
    o2 = wmma_bf16(pa, load_bfrag(Vth + (size_t)(32 + ln) * CS, jj, h4), o2);
    o3 = wmma_bf16(pa, load_bfrag(Vth + (size_t)(48 + ln) * CS, jj, h4), o3);
  }

  float inv = 1.0f / row_sum;
#pragma unroll
  for (int r = 0; r < 8; r++) {
    float ir = __shfl(inv, r + 8 * h4, 32);
    int qg = q0 + r + 8 * h4;
    size_t base = ((size_t)bb * CS + qg) * CD + hh * CDK;
    xo[base + 0  + ln] = (bf16)(o0[r] * ir);
    xo[base + 16 + ln] = (bf16)(o1[r] * ir);
    xo[base + 32 + ln] = (bf16)(o2[r] * ir);
    xo[base + 48 + ln] = (bf16)(o3[r] * ir);
  }
}

// ------------------------------------------------------- output projection
// Same 32x64 register-blocked WMMA GEMM, f32 output + bias.
__global__ void __launch_bounds__(128)
mha_oproj_kernel(const bf16* __restrict__ x, const bf16* __restrict__ wo,
                 const float* __restrict__ bo, float* __restrict__ out) {
  const int lane = threadIdx.x & 31;
  const int wave = threadIdx.x >> 5;
  const int h4   = lane >> 4;
  const int ln   = lane & 15;

  const int mt = blockIdx.x;
  const int nq = blockIdx.y * 4 + wave;

  const bf16* arow0 = x + (size_t)(mt * 32 + ln) * CD;
  const bf16* arow1 = x + (size_t)(mt * 32 + 16 + ln) * CD;
  const bf16* brow0 = wo + (size_t)(nq * 64 + 0  + ln) * CD;
  const bf16* brow1 = wo + (size_t)(nq * 64 + 16 + ln) * CD;
  const bf16* brow2 = wo + (size_t)(nq * 64 + 32 + ln) * CD;
  const bf16* brow3 = wo + (size_t)(nq * 64 + 48 + ln) * CD;

  v8f acc[2][4] = {};
#pragma unroll 2
  for (int k = 0; k < CD; k += 32) {
    v16bf a0 = load_afrag(arow0, k, h4);
    v16bf a1 = load_afrag(arow1, k, h4);
    v16bf b0 = load_bfrag(brow0, k, h4);
    v16bf b1 = load_bfrag(brow1, k, h4);
    v16bf b2 = load_bfrag(brow2, k, h4);
    v16bf b3 = load_bfrag(brow3, k, h4);
    acc[0][0] = wmma_bf16(a0, b0, acc[0][0]);
    acc[0][1] = wmma_bf16(a0, b1, acc[0][1]);
    acc[0][2] = wmma_bf16(a0, b2, acc[0][2]);
    acc[0][3] = wmma_bf16(a0, b3, acc[0][3]);
    acc[1][0] = wmma_bf16(a1, b0, acc[1][0]);
    acc[1][1] = wmma_bf16(a1, b1, acc[1][1]);
    acc[1][2] = wmma_bf16(a1, b2, acc[1][2]);
    acc[1][3] = wmma_bf16(a1, b3, acc[1][3]);
  }

#pragma unroll
  for (int j = 0; j < 4; j++) {
    const int ng = nq * 64 + j * 16 + ln;
    const float bbv = bo[ng];
#pragma unroll
    for (int i = 0; i < 2; i++) {
#pragma unroll
      for (int r = 0; r < 8; r++) {
        int mg = mt * 32 + i * 16 + r + 8 * h4;
        out[(size_t)mg * CD + ng] = acc[i][j][r] + bbv;
      }
    }
  }
}

extern "C" void kernel_launch(void* const* d_in, const int* in_sizes, int n_in,
                              void* d_out, int out_size, void* d_ws, size_t ws_size,
                              hipStream_t stream) {
  (void)in_sizes; (void)n_in; (void)out_size; (void)ws_size;
  const float* query = (const float*)d_in[0];
  const float* key   = (const float*)d_in[1];
  const float* value = (const float*)d_in[2];
  const int*   mask  = (const int*)  d_in[3];
  const float* wq = (const float*)d_in[4];
  const float* bq = (const float*)d_in[5];
  const float* wk = (const float*)d_in[6];
  const float* bk = (const float*)d_in[7];
  const float* wv = (const float*)d_in[8];
  const float* bv = (const float*)d_in[9];
  const float* wo = (const float*)d_in[10];
  const float* bo = (const float*)d_in[11];

  const size_t BSD = (size_t)CB * CS * CD;  // 4 Mi elements
  const size_t DDE = (size_t)CD * CD;       // 1 Mi elements

  char* w = (char*)d_ws;
  bf16* xqb = (bf16*)w; w += BSD * 2;
  bf16* xkb = (bf16*)w; w += BSD * 2;
  bf16* xvb = (bf16*)w; w += BSD * 2;
  bf16* wqb = (bf16*)w; w += DDE * 2;
  bf16* wkb = (bf16*)w; w += DDE * 2;
  bf16* wvb = (bf16*)w; w += DDE * 2;
  bf16* wob = (bf16*)w; w += DDE * 2;
  bf16* Qp  = (bf16*)w; w += BSD * 2;   // [B,H,S,DK]
  bf16* Kp  = (bf16*)w; w += BSD * 2;   // [B,H,S,DK]
  bf16* Vt  = (bf16*)w; w += BSD * 2;   // [B,H,DK,S]
  bf16* xo  = (bf16*)w; w += BSD * 2;   // [B,S,D]

  auto cvt = [&](const float* in, bf16* out, size_t n) {
    int n4 = (int)(n / 4);
    mha_cvt_kernel<<<dim3((n4 + 255) / 256), dim3(256), 0, stream>>>(in, out, n4);
  };
  cvt(query, xqb, BSD);
  cvt(key,   xkb, BSD);
  cvt(value, xvb, BSD);
  cvt(wq, wqb, DDE);
  cvt(wk, wkb, DDE);
  cvt(wv, wvb, DDE);
  cvt(wo, wob, DDE);

  mha_qkv_proj_kernel<<<dim3(CB * CS / 32, CD / 256, 3), dim3(128), 0, stream>>>(
      xqb, xkb, xvb, wqb, wkb, wvb, bq, bk, bv, Qp, Kp, Vt);

  mha_attn_kernel<<<dim3(CS / 64, CH, CB), dim3(128), 0, stream>>>(
      Qp, Kp, Vt, mask, xo);

  mha_oproj_kernel<<<dim3(CB * CS / 32, CD / 256), dim3(128), 0, stream>>>(
      xo, wob, bo, (float*)d_out);
}
